// ZOCA_72533407694904
// MI455X (gfx1250) — matrix operand order repaired and checked by reading, not compiled, not run on hardware
//
#include <hip/hip_runtime.h>

#define DEV __device__ __forceinline__

constexpr int B = 16, T = 128, D = 384, H = 6, DH = 64, N = B * T;

typedef __attribute__((ext_vector_type(16))) __bf16 v16bf;
typedef __attribute__((ext_vector_type(8)))  __bf16 v8bf;
typedef __attribute__((ext_vector_type(8)))  float  v8f;

// ---------- helpers ----------

DEV unsigned short f2bf(float f) {
  union { float f; unsigned u; } x; x.f = f;
  unsigned u = x.u;
  u += 0x7fffu + ((u >> 16) & 1u);            // round-to-nearest-even
  return (unsigned short)(u >> 16);
}

DEV v8f wmma_bf16(v16bf a, v16bf b, v8f c) {
  // D = A(16x32) * B(32x16) + C ; f32 accumulate
  return __builtin_amdgcn_wmma_f32_16x16x32_bf16(false, a, false, b, (short)0, c,
                                                 false, false);
}

// A fragment (16x32 bf16, row-major source): lane m=lane&15 half=lane>>4
// half 0 holds K = 0..15, half 1 holds K = 16..31
DEV v16bf load_a_frag(const unsigned short* base, int ld, int kb, int lane) {
  int m = lane & 15, half = lane >> 4;
  const unsigned short* p = base + m * ld + kb + half * 16;
  union { v16bf v; v8bf h[2]; } f;
  f.h[0] = *(const v8bf*)(p);
  f.h[1] = *(const v8bf*)(p + 8);
  return f.v;
}

// B fragment (32x16 bf16): column n of B = row n of "Bsrc" (contiguous over K).
// lanes 0-15 hold K=0..15, lanes 16-31 hold K=16..31 (contiguous 32B load)
DEV v16bf load_b_frag(const unsigned short* base, int ld, int kb, int lane) {
  int n = lane & 15, half = lane >> 4;
  return *(const v16bf*)(base + n * ld + kb + half * 16);
}

DEV v16bf neg_frag(v16bf x) {
  union { v16bf v; unsigned short u[16]; } f; f.v = x;
#pragma unroll
  for (int i = 0; i < 16; ++i) f.u[i] ^= 0x8000u;
  return f.v;
}

DEV float block_sum_128(float v, float* sb) {  // blockDim.x == 128 (4 waves)
#pragma unroll
  for (int off = 16; off > 0; off >>= 1) v += __shfl_xor(v, off, 32);
  int w = threadIdx.x >> 5;
  if ((threadIdx.x & 31) == 0) sb[w] = v;
  __syncthreads();
  float r = sb[0] + sb[1] + sb[2] + sb[3];
  __syncthreads();
  return r;
}

// ---------- kernels ----------

__global__ void cvt_bf16(const float* __restrict__ src,
                         unsigned short* __restrict__ dst, int n) {
  int i = blockIdx.x * 256 + threadIdx.x;
  if (i < n) dst[i] = f2bf(src[i]);
}

__global__ void compute_ih(const float* __restrict__ iproj, float* __restrict__ ih) {
  int b = blockIdx.x;                          // 128 threads, D/3 == 128
  const float* p = iproj + b * (D / 3);
  __shared__ float sb[4];
  float t = p[threadIdx.x];
  float ss = block_sum_128(t * t, sb);
  float inv = rsqrtf(3.0f * ss);               // norm3 cancels algebraically
#pragma unroll
  for (int d = threadIdx.x; d < D; d += 128)
    ih[b * D + d] = p[d & 127] * inv;
}

// C(N x D) = Xbf(N x D) @ Wbf.T(D x D) + bias  (fp32 out)
__global__ __launch_bounds__(256)
void gemm_xwT(const unsigned short* __restrict__ X, const unsigned short* __restrict__ W,
              const float* __restrict__ bias, float* __restrict__ C) {
  int lane = threadIdx.x & 31, wv = threadIdx.x >> 5;
  int m0 = blockIdx.y * 128 + wv * 16;
  int c0 = blockIdx.x * 64;
  const unsigned short* Abase = X + (size_t)m0 * D;
  v8f acc[4] = {};
  for (int kk = 0; kk < D; kk += 32) {
    v16bf a = load_a_frag(Abase, D, kk, lane);
#pragma unroll
    for (int t = 0; t < 4; ++t) {
      v16bf b = load_b_frag(W + (size_t)(c0 + 16 * t) * D, D, kk, lane);
      acc[t] = wmma_bf16(a, b, acc[t]);
    }
  }
  int half = lane >> 4, nl = lane & 15;
#pragma unroll
  for (int t = 0; t < 4; ++t)
#pragma unroll
    for (int j = 0; j < 8; ++j) {
      int row = m0 + half * 8 + j, col = c0 + 16 * t + nl;
      C[(size_t)row * D + col] = acc[t][j] + bias[col];
    }
}

// q -= (q . ih_b) * ih_b ; write bf16
__global__ void project_ih(const float* __restrict__ Qin, const float* __restrict__ ih,
                           unsigned short* __restrict__ Qout) {
  int r = blockIdx.x, b = r / T;
  const float* q = Qin + (size_t)r * D;
  const float* v = ih + (size_t)b * D;
  __shared__ float sb[4];
  float p = 0.f;
  for (int d = threadIdx.x; d < D; d += 128) p += q[d] * v[d];
  float dot = block_sum_128(p, sb);
  for (int d = threadIdx.x; d < D; d += 128)
    Qout[(size_t)r * D + d] = f2bf(q[d] - dot * v[d]);
}

// Vt[b][d][t] = bf16(V[b][t][d])
__global__ void vtrans(const float* __restrict__ V, unsigned short* __restrict__ Vt) {
  int i = blockIdx.x * 256 + threadIdx.x;
  if (i >= N * D) return;
  int d = i % D, t = (i / D) % T, b = i / (D * T);
  Vt[((size_t)b * D + d) * T + t] = f2bf(V[i]);
}

// scores = Qh @ Kh.T * DH^-0.5, softmax over k; write fp32 to 2 output groups
// and bf16 to workspace. grid = (4 combos, H, B), 8 waves of 16 q-rows.
__global__ __launch_bounds__(256)
void attn_scores_softmax(const unsigned short* __restrict__ Qpr,
                         const unsigned short* __restrict__ Qpi,
                         const unsigned short* __restrict__ Kpr,
                         const unsigned short* __restrict__ Kpi,
                         unsigned short* __restrict__ w0, unsigned short* __restrict__ w1,
                         unsigned short* __restrict__ w2, unsigned short* __restrict__ w3,
                         float* __restrict__ outw) {
  int c = blockIdx.x, h = blockIdx.y, b = blockIdx.z;
  int lane = threadIdx.x & 31, wv = threadIdx.x >> 5;
  const unsigned short* Q = (c < 2) ? Qpr : Qpi;
  const unsigned short* K = (c & 1) ? Kpi : Kpr;
  unsigned short* wc = (c == 0) ? w0 : (c == 1) ? w1 : (c == 2) ? w2 : w3;

  const unsigned short* Abase = Q + ((size_t)(b * T + wv * 16)) * D + h * DH;
  v8f acc[8] = {};
#pragma unroll
  for (int kk = 0; kk < DH; kk += 32) {
    v16bf a = load_a_frag(Abase, D, kk, lane);
#pragma unroll
    for (int t = 0; t < 8; ++t) {
      v16bf bb = load_b_frag(K + ((size_t)(b * T + 16 * t)) * D + h * DH, D, kk, lane);
      acc[t] = wmma_bf16(a, bb, acc[t]);
    }
  }
  const float scale = 0.125f;  // 64^-0.5
  int half = lane >> 4, nl = lane & 15;
#pragma unroll
  for (int j = 0; j < 8; ++j) {          // per row held by this half-wave
    float mx = -3.4e38f;
#pragma unroll
    for (int t = 0; t < 8; ++t) { acc[t][j] *= scale; mx = fmaxf(mx, acc[t][j]); }
#pragma unroll
    for (int off = 1; off < 16; off <<= 1) mx = fmaxf(mx, __shfl_xor(mx, off, 32));
    float sum = 0.f;
#pragma unroll
    for (int t = 0; t < 8; ++t) { float e = __expf(acc[t][j] - mx); acc[t][j] = e; sum += e; }
#pragma unroll
    for (int off = 1; off < 16; off <<= 1) sum += __shfl_xor(sum, off, 32);
    float inv = 1.0f / sum;
#pragma unroll
    for (int t = 0; t < 8; ++t) acc[t][j] *= inv;
  }
  // combo -> output groups: rr:{0,1} ri:{2,4} ir:{3,5} ii:{6,7}
  const int gA[4] = {0, 2, 3, 6}, gB[4] = {1, 4, 5, 7};
  int q0 = wv * 16 + half * 8;
#pragma unroll
  for (int t = 0; t < 8; ++t)
#pragma unroll
    for (int j = 0; j < 8; ++j) {
      int q = q0 + j, k = 16 * t + nl;
      float val = acc[t][j];
      size_t base = (size_t)h * T * T + (size_t)q * T + k;
      outw[((size_t)(gA[c] * B + b)) * H * T * T + base] = val;
      outw[((size_t)(gB[c] * B + b)) * H * T * T + base] = val;
      wc[((size_t)(b * H + h) * T + q) * T + k] = f2bf(val);
    }
}

// P_re = A_rr Vr - A_ri Vi - A_ir Vi - A_ii Vr ; P_im = A_rr Vi + A_ir Vr + A_ri Vr - A_ii Vi
__global__ __launch_bounds__(256)
void attn_av(const unsigned short* __restrict__ w0, const unsigned short* __restrict__ w1,
             const unsigned short* __restrict__ w2, const unsigned short* __restrict__ w3,
             const unsigned short* __restrict__ Vtr, const unsigned short* __restrict__ Vti,
             unsigned short* __restrict__ Pre, unsigned short* __restrict__ Pim) {
  int h = blockIdx.x, b = blockIdx.y;
  int lane = threadIdx.x & 31, wv = threadIdx.x >> 5;
  size_t arow = ((size_t)(b * H + h) * T + wv * 16) * T;
  const unsigned short* Arr = w0 + arow;
  const unsigned short* Ari = w1 + arow;
  const unsigned short* Air = w2 + arow;
  const unsigned short* Aii = w3 + arow;
  v8f are[4] = {}, aim[4] = {};
  for (int kk = 0; kk < T; kk += 32) {
    v16bf a_rr = load_a_frag(Arr, T, kk, lane);
    v16bf a_ri = load_a_frag(Ari, T, kk, lane);
    v16bf a_ir = load_a_frag(Air, T, kk, lane);
    v16bf a_ii = load_a_frag(Aii, T, kk, lane);
#pragma unroll
    for (int t = 0; t < 4; ++t) {
      v16bf bvr = load_b_frag(Vtr + ((size_t)b * D + h * DH + 16 * t) * T, T, kk, lane);
      v16bf bvi = load_b_frag(Vti + ((size_t)b * D + h * DH + 16 * t) * T, T, kk, lane);
      v16bf nvr = neg_frag(bvr), nvi = neg_frag(bvi);
      are[t] = wmma_bf16(a_rr, bvr, are[t]);
      are[t] = wmma_bf16(a_ri, nvi, are[t]);
      are[t] = wmma_bf16(a_ir, nvi, are[t]);
      are[t] = wmma_bf16(a_ii, nvr, are[t]);
      aim[t] = wmma_bf16(a_rr, bvi, aim[t]);
      aim[t] = wmma_bf16(a_ir, bvr, aim[t]);
      aim[t] = wmma_bf16(a_ri, bvr, aim[t]);
      aim[t] = wmma_bf16(a_ii, nvi, aim[t]);
    }
  }
  int half = lane >> 4, nl = lane & 15;
#pragma unroll
  for (int t = 0; t < 4; ++t)
#pragma unroll
    for (int j = 0; j < 8; ++j) {
      int row = b * T + wv * 16 + half * 8 + j;
      int col = h * DH + 16 * t + nl;
      Pre[(size_t)row * D + col] = f2bf(are[t][j]);
      Pim[(size_t)row * D + col] = f2bf(aim[t][j]);
    }
}

// X_re = Pre @ Wo.T - 2*bo ; X_im = Pim @ Wo.T + 2*bo  (bf16 out)
__global__ __launch_bounds__(256)
void gemm_wo(const unsigned short* __restrict__ Pre, const unsigned short* __restrict__ Pim,
             const unsigned short* __restrict__ Wo_, const float* __restrict__ bo,
             unsigned short* __restrict__ Xre, unsigned short* __restrict__ Xim) {
  int lane = threadIdx.x & 31, wv = threadIdx.x >> 5;
  int m0 = blockIdx.y * 128 + wv * 16;
  int c0 = blockIdx.x * 64;
  v8f ar[4] = {}, ai[4] = {};
  for (int kk = 0; kk < D; kk += 32) {
    v16bf a_re = load_a_frag(Pre + (size_t)m0 * D, D, kk, lane);
    v16bf a_im = load_a_frag(Pim + (size_t)m0 * D, D, kk, lane);
#pragma unroll
    for (int t = 0; t < 4; ++t) {
      v16bf b = load_b_frag(Wo_ + (size_t)(c0 + 16 * t) * D, D, kk, lane);
      ar[t] = wmma_bf16(a_re, b, ar[t]);
      ai[t] = wmma_bf16(a_im, b, ai[t]);
    }
  }
  int half = lane >> 4, nl = lane & 15;
#pragma unroll
  for (int t = 0; t < 4; ++t)
#pragma unroll
    for (int j = 0; j < 8; ++j) {
      int row = m0 + half * 8 + j, col = c0 + 16 * t + nl;
      Xre[(size_t)row * D + col] = f2bf(ar[t][j] - 2.0f * bo[col]);
      Xim[(size_t)row * D + col] = f2bf(ai[t][j] + 2.0f * bo[col]);
    }
}

// G_re = gelu(Xre@W1r.T + b1r - Xim@W1i.T - b1i) ; G_im = gelu(Xim@W1r.T + b1r + Xre@W1i.T + b1i)
__global__ __launch_bounds__(256)
void gemm_w1_gelu(const unsigned short* __restrict__ Xre, const unsigned short* __restrict__ Xim,
                  const unsigned short* __restrict__ W1r_, const unsigned short* __restrict__ W1i_,
                  const float* __restrict__ b1r, const float* __restrict__ b1i,
                  float* __restrict__ Gre, float* __restrict__ Gim) {
  int lane = threadIdx.x & 31, wv = threadIdx.x >> 5;
  int m0 = blockIdx.y * 128 + wv * 16;
  int c0 = blockIdx.x * 64;
  v8f ar[4] = {}, ai[4] = {};
  for (int kk = 0; kk < D; kk += 32) {
    v16bf a_re = load_a_frag(Xre + (size_t)m0 * D, D, kk, lane);
    v16bf a_im = load_a_frag(Xim + (size_t)m0 * D, D, kk, lane);
#pragma unroll
    for (int t = 0; t < 4; ++t) {
      v16bf b_r = load_b_frag(W1r_ + (size_t)(c0 + 16 * t) * D, D, kk, lane);
      v16bf b_i = load_b_frag(W1i_ + (size_t)(c0 + 16 * t) * D, D, kk, lane);
      v16bf b_in = neg_frag(b_i);
      ar[t] = wmma_bf16(a_re, b_r, ar[t]);
      ar[t] = wmma_bf16(a_im, b_in, ar[t]);
      ai[t] = wmma_bf16(a_im, b_r, ai[t]);
      ai[t] = wmma_bf16(a_re, b_i, ai[t]);
    }
  }
  int half = lane >> 4, nl = lane & 15;
#pragma unroll
  for (int t = 0; t < 4; ++t)
#pragma unroll
    for (int j = 0; j < 8; ++j) {
      int row = m0 + half * 8 + j, col = c0 + 16 * t + nl;
      float re = ar[t][j] + b1r[col] - b1i[col];
      float im = ai[t][j] + b1r[col] + b1i[col];
      re = 0.5f * re * (1.0f + erff(re * 0.70710678118f));
      im = 0.5f * im * (1.0f + erff(im * 0.70710678118f));
      Gre[(size_t)row * D + col] = re;
      Gim[(size_t)row * D + col] = im;
    }
}

// W2 dots, complex-scalar algebra, layernorm; writes X_r, X_i regions of d_out
__global__ void finalize(const float* __restrict__ Gre, const float* __restrict__ Gim,
                         const float* __restrict__ W2r, const float* __restrict__ W2i,
                         const float* __restrict__ b2r, const float* __restrict__ b2i,
                         const float* __restrict__ real, const float* __restrict__ imag,
                         const float* __restrict__ g1, const float* __restrict__ be1,
                         float* __restrict__ out) {
  int r = blockIdx.x;
  __shared__ float sb[4];
  const float* gr = Gre + (size_t)r * D;
  const float* gi = Gim + (size_t)r * D;
  const float* re = real + (size_t)r * D;
  const float* im = imag + (size_t)r * D;
  float p_rr = 0, p_ri = 0, p_ir = 0, p_ii = 0, p_fr = 0, p_fi = 0;
  for (int d = threadIdx.x; d < D; d += 128) {
    float wr = W2r[d], wi = W2i[d];
    p_rr += gr[d] * wr; p_ri += gr[d] * wi;
    p_ir += gi[d] * wr; p_ii += gi[d] * wi;
    float a = re[d], bb = im[d];
    p_fr += a * a + bb * bb;
    p_fi += -a * bb - bb * bb;
  }
  float dot_rr = block_sum_128(p_rr, sb);
  float dot_ri = block_sum_128(p_ri, sb);
  float dot_ir = block_sum_128(p_ir, sb);
  float dot_ii = block_sum_128(p_ii, sb);
  float F_re = block_sum_128(p_fr, sb);
  float F_im = block_sum_128(p_fi, sb);
  float sre = dot_rr + b2r[0] - dot_ii - b2i[0];
  float sim = dot_ir + b2r[0] + dot_ri + b2i[0];
  float Sre = sre * sre + sim * sim;           // S_re' = s_re^2 + s_im^2
  float Sim = -Sre * sim - sim * sim;          // uses updated S_re (sequential semantics)
  float denom = F_re * F_re + F_im * F_im;
  float wvv = (Sre * F_re + Sim * F_im) / denom;   // W_re == W_im
  float sxr = 0, sxr2 = 0, sxi = 0, sxi2 = 0;
  for (int d = threadIdx.x; d < D; d += 128) {
    float xr = wvv * (re[d] - im[d]);
    float xi = 2.0f * wvv * im[d];
    sxr += xr; sxr2 += xr * xr; sxi += xi; sxi2 += xi * xi;
  }
  float mr = block_sum_128(sxr, sb) / D;
  float vr = block_sum_128(sxr2, sb) / D - mr * mr;
  float mi = block_sum_128(sxi, sb) / D;
  float vi = block_sum_128(sxi2, sb) / D - mi * mi;
  float irn = rsqrtf(vr + 1e-5f), iin = rsqrtf(vi + 1e-5f);
  float* Xr = out;
  float* Xi = out + (size_t)N * D;
  for (int d = threadIdx.x; d < D; d += 128) {
    float xr = wvv * (re[d] - im[d]);
    float xi = 2.0f * wvv * im[d];
    Xr[(size_t)r * D + d] = (xr - mr) * irn * g1[d] + be1[d];
    Xi[(size_t)r * D + d] = (xi - mi) * iin * g1[d] + be1[d];
  }
}

// ---------- host ----------

extern "C" void kernel_launch(void* const* d_in, const int* in_sizes, int n_in,
                              void* d_out, int out_size, void* d_ws, size_t ws_size,
                              hipStream_t stream) {
  (void)in_sizes; (void)n_in; (void)out_size; (void)ws_size;
  const float* real  = (const float*)d_in[0];
  const float* imag  = (const float*)d_in[1];
  const float* iproj = (const float*)d_in[2];
  const float* Wq = (const float*)d_in[3];  const float* bq = (const float*)d_in[4];
  const float* Wk = (const float*)d_in[5];  const float* bk = (const float*)d_in[6];
  const float* Wv = (const float*)d_in[7];  const float* bv = (const float*)d_in[8];
  const float* Wo = (const float*)d_in[9];  const float* bo = (const float*)d_in[10];
  const float* W1r = (const float*)d_in[11]; const float* b1r = (const float*)d_in[12];
  const float* W1i = (const float*)d_in[13]; const float* b1i = (const float*)d_in[14];
  const float* W2r = (const float*)d_in[15]; const float* b2r = (const float*)d_in[16];
  const float* W2i = (const float*)d_in[17]; const float* b2i = (const float*)d_in[18];
  const float* g1  = (const float*)d_in[19]; const float* be1 = (const float*)d_in[20];
  float* out = (float*)d_out;

  char* ws = (char*)d_ws;
  size_t off = 0;
  auto alloc = [&](size_t bytes) -> void* {
    void* p = ws + off;
    off = (off + bytes + 255) & ~(size_t)255;
    return p;
  };
  const size_t ND = (size_t)N * D, DD = (size_t)D * D, WW = (size_t)B * H * T * T;
  unsigned short* real_bf = (unsigned short*)alloc(ND * 2);
  unsigned short* imag_bf = (unsigned short*)alloc(ND * 2);
  unsigned short* wq_bf  = (unsigned short*)alloc(DD * 2);
  unsigned short* wk_bf  = (unsigned short*)alloc(DD * 2);
  unsigned short* wv_bf  = (unsigned short*)alloc(DD * 2);
  unsigned short* wo_bf  = (unsigned short*)alloc(DD * 2);
  unsigned short* w1r_bf = (unsigned short*)alloc(DD * 2);
  unsigned short* w1i_bf = (unsigned short*)alloc(DD * 2);
  float* ih = (float*)alloc((size_t)B * D * 4);
  float* Qr = (float*)alloc(ND * 4); float* Qi = (float*)alloc(ND * 4);
  float* Kr = (float*)alloc(ND * 4); float* Ki = (float*)alloc(ND * 4);
  float* Vr = (float*)alloc(ND * 4); float* Vi = (float*)alloc(ND * 4);
  unsigned short* Qpr = (unsigned short*)alloc(ND * 2);
  unsigned short* Qpi = (unsigned short*)alloc(ND * 2);
  unsigned short* Kpr = (unsigned short*)alloc(ND * 2);
  unsigned short* Kpi = (unsigned short*)alloc(ND * 2);
  unsigned short* Vtr = (unsigned short*)alloc(ND * 2);
  unsigned short* Vti = (unsigned short*)alloc(ND * 2);
  unsigned short* w_rr = (unsigned short*)alloc(WW * 2);
  unsigned short* w_ri = (unsigned short*)alloc(WW * 2);
  unsigned short* w_ir = (unsigned short*)alloc(WW * 2);
  unsigned short* w_ii = (unsigned short*)alloc(WW * 2);
  unsigned short* Pre = (unsigned short*)alloc(ND * 2);
  unsigned short* Pim = (unsigned short*)alloc(ND * 2);
  unsigned short* Xre = (unsigned short*)alloc(ND * 2);
  unsigned short* Xim = (unsigned short*)alloc(ND * 2);
  float* Gre = (float*)alloc(ND * 4);
  float* Gim = (float*)alloc(ND * 4);

  // converts
  cvt_bf16<<<(int)((ND + 255) / 256), 256, 0, stream>>>(real, real_bf, (int)ND);
  cvt_bf16<<<(int)((ND + 255) / 256), 256, 0, stream>>>(imag, imag_bf, (int)ND);
  cvt_bf16<<<(int)((DD + 255) / 256), 256, 0, stream>>>(Wq, wq_bf, (int)DD);
  cvt_bf16<<<(int)((DD + 255) / 256), 256, 0, stream>>>(Wk, wk_bf, (int)DD);
  cvt_bf16<<<(int)((DD + 255) / 256), 256, 0, stream>>>(Wv, wv_bf, (int)DD);
  cvt_bf16<<<(int)((DD + 255) / 256), 256, 0, stream>>>(Wo, wo_bf, (int)DD);
  cvt_bf16<<<(int)((DD + 255) / 256), 256, 0, stream>>>(W1r, w1r_bf, (int)DD);
  cvt_bf16<<<(int)((DD + 255) / 256), 256, 0, stream>>>(W1i, w1i_bf, (int)DD);

  compute_ih<<<B, 128, 0, stream>>>(iproj, ih);

  dim3 gg(D / 64, N / 128);
  gemm_xwT<<<gg, 256, 0, stream>>>(real_bf, wq_bf, bq, Qr);
  gemm_xwT<<<gg, 256, 0, stream>>>(imag_bf, wq_bf, bq, Qi);
  gemm_xwT<<<gg, 256, 0, stream>>>(real_bf, wk_bf, bk, Kr);
  gemm_xwT<<<gg, 256, 0, stream>>>(imag_bf, wk_bf, bk, Ki);
  gemm_xwT<<<gg, 256, 0, stream>>>(real_bf, wv_bf, bv, Vr);
  gemm_xwT<<<gg, 256, 0, stream>>>(imag_bf, wv_bf, bv, Vi);

  project_ih<<<N, 128, 0, stream>>>(Qr, ih, Qpr);
  project_ih<<<N, 128, 0, stream>>>(Qi, ih, Qpi);
  project_ih<<<N, 128, 0, stream>>>(Kr, ih, Kpr);
  project_ih<<<N, 128, 0, stream>>>(Ki, ih, Kpi);

  vtrans<<<(int)((ND + 255) / 256), 256, 0, stream>>>(Vr, Vtr);
  vtrans<<<(int)((ND + 255) / 256), 256, 0, stream>>>(Vi, Vti);

  float* outw = out + 2 * ND;  // weights region of d_out
  attn_scores_softmax<<<dim3(4, H, B), 256, 0, stream>>>(Qpr, Qpi, Kpr, Kpi,
                                                         w_rr, w_ri, w_ir, w_ii, outw);

  attn_av<<<dim3(H, B), 256, 0, stream>>>(w_rr, w_ri, w_ir, w_ii, Vtr, Vti, Pre, Pim);

  gemm_wo<<<gg, 256, 0, stream>>>(Pre, Pim, wo_bf, bo, Xre, Xim);

  gemm_w1_gelu<<<gg, 256, 0, stream>>>(Xre, Xim, w1r_bf, w1i_bf, b1r, b1i, Gre, Gim);

  finalize<<<N, 128, 0, stream>>>(Gre, Gim, W2r, W2i, b2r, b2i, real, imag, g1, be1, out);
}
